// MambaBlock_4793183502527
// MI455X (gfx1250) — matrix operand order repaired and checked by reading, not compiled, not run on hardware
//
#include <hip/hip_runtime.h>
#include <hip/hip_bf16.h>

// ---------------------------------------------------------------------------
// Mamba block for MI455X (gfx1250):
//   - WMMA-f16 GEMMs with async-to-LDS double-buffered weight tiles
//   - register-resident SSM scan (h[64] per thread, B/C via LDS)
// ---------------------------------------------------------------------------

typedef __attribute__((ext_vector_type(16))) _Float16 v16h;
typedef __attribute__((ext_vector_type(8)))  _Float16 v8h;
typedef __attribute__((ext_vector_type(8)))  float    v8f;

constexpr int BATCH = 4;
constexpr int SEQ   = 2048;
constexpr int DIM   = 1024;
constexpr int DI    = 2 * DIM;       // 2048
constexpr int NST   = 64;
constexpr int DTR   = DIM / 16;      // 64
constexpr int DCONV = 4;
constexpr int BL    = BATCH * SEQ;   // 8192 rows

#if __has_builtin(__builtin_amdgcn_exp2f)
#define EXP2F(x) __builtin_amdgcn_exp2f(x)
#else
#define EXP2F(x) exp2f(x)
#endif

// Async copy 16 bytes global -> LDS (per-lane), tracked by ASYNCcnt.
__device__ __forceinline__ void async_copy_b128(unsigned lds_addr, const void* gaddr) {
  asm volatile("global_load_async_to_lds_b128 %0, %1, off"
               :: "v"(lds_addr), "v"(gaddr)
               : "memory");
}
__device__ __forceinline__ void wait_async_le2() {
  asm volatile("s_wait_asynccnt 2" ::: "memory");
}
__device__ __forceinline__ void wait_async_le0() {
  asm volatile("s_wait_asynccnt 0" ::: "memory");
}

// ---------------------------------------------------------------------------
// WMMA GEMM:  C[M x N] (f32) = A[M x K] (f16 row-major) @ Bt[N x K] (f16,
// weights pre-transposed).  Block = 128 threads = 4 waves; block tile
// 64(M) x 64(N); wave tile 16 x 64.  The 64x32 weight tile is block-uniform,
// so it is staged once per block into LDS via global_load_async_to_lds_b128
// (double buffered), and all 4 waves read fragments with ds loads.
//
// Fragment layouts used by this code (wave32):
//   A 16x32 f16 : lane m=lane&15, half=lane>>4; slot i -> K=half*16+i
//   B 32x16 f16 : lane n=lane&15, half=lane>>4; slot i -> K=half*16+i
//   C/D 16x16 f32: VGPR r, lane: M = r + 8*(lane>>4), N = lane&15
// ---------------------------------------------------------------------------
__global__ __launch_bounds__(128)
void gemm_wmma_f16(const _Float16* __restrict__ A,
                   const _Float16* __restrict__ Bt,
                   const float*    __restrict__ residual,   // nullptr -> none
                   float*          __restrict__ C,
                   int M, int N, int K) {
  __shared__ _Float16 sB[2 * 64 * 32];   // double-buffered 64(N) x 32(K) tile

  const int tid  = threadIdx.x;
  const int lane = tid & 31;
  const int wave = tid >> 5;
  const int hl   = lane >> 4;      // lane half (0/1)
  const int l16  = lane & 15;
  const int m0   = blockIdx.y * 64 + wave * 16;
  const int n0   = blockIdx.x * 64;
  const unsigned lds0 = (unsigned)(uintptr_t)(void*)sB;  // LDS byte offset

  const int ntiles = K >> 5;       // K is a multiple of 32 (>= 64)

  // Cooperative async stage of one 64(N) x 32(K) f16 tile (4 KB):
  // 256 x 16B chunks, 128 threads -> 2 async b128 per thread (2 per wave).
  auto issue_tile = [&](int t, int buf) {
    const int kk = t << 5;
#pragma unroll
    for (int c = 0; c < 2; ++c) {
      const int chunk = tid + c * 128;
      const int row = chunk >> 2;          // 0..63 (N within tile)
      const int seg = chunk & 3;           // 16B segment within 64B row
      const _Float16* g = Bt + (size_t)(n0 + row) * K + kk + seg * 8;
      async_copy_b128(lds0 + (unsigned)(buf * 4096 + row * 64 + seg * 16), g);
    }
  };

  issue_tile(0, 0);
  issue_tile(1, 1);

  v8f acc[4] = {v8f{}, v8f{}, v8f{}, v8f{}};
  const _Float16* Arow = A + (size_t)(m0 + l16) * K;

  for (int t = 0; t < ntiles; ++t) {
    // Oldest in-flight tile (2 async ops per wave) must be complete.
    if (t + 1 < ntiles) wait_async_le2();
    else                wait_async_le0();
    __syncthreads();                       // all waves' chunks visible

    const int k = t << 5;
    if (k + 128 < K) {
      __builtin_prefetch((const void*)(Arow + k + 128), 0, 0);  // global_prefetch_b8
    }
    // A fragment: two contiguous 8-half chunks from global
    v8h alo = *(const v8h*)(Arow + k + hl * 16);
    v8h ahi = *(const v8h*)(Arow + k + hl * 16 + 8);
    v16h a;
#pragma unroll
    for (int i = 0; i < 8; ++i) { a[i] = alo[i]; a[8 + i] = ahi[i]; }

    const _Float16* sbuf = sB + (t & 1) * 2048;
#pragma unroll
    for (int tt = 0; tt < 4; ++tt) {
      v16h b = *(const v16h*)(sbuf + (tt * 16 + l16) * 32 + hl * 16);  // ds load
      acc[tt] = __builtin_amdgcn_wmma_f32_16x16x32_f16(
          /*neg_a=*/false, a, /*neg_b=*/false, b,
          /*c_mod=*/(short)0, acc[tt], /*reuse_a=*/false, /*reuse_b=*/false);
    }
    __syncthreads();                       // everyone done reading buf (t&1)
    if (t + 2 < ntiles) issue_tile(t + 2, t & 1);
  }

#pragma unroll
  for (int tt = 0; tt < 4; ++tt) {
#pragma unroll
    for (int r = 0; r < 8; ++r) {
      const int m = m0 + hl * 8 + r;
      const int n = n0 + tt * 16 + l16;
      const size_t idx = (size_t)m * N + n;
      float v = acc[tt][r];
      if (residual) v += residual[idx];
      C[idx] = v;
    }
  }
}

// ---------------------------------------------------------------------------
// Weight prep: W[K x N] f32  ->  Wt[N x K] f16
// ---------------------------------------------------------------------------
__global__ __launch_bounds__(256)
void transpose_cvt_f16(const float* __restrict__ W, _Float16* __restrict__ Wt,
                       int K, int N) {
  size_t idx = (size_t)blockIdx.x * blockDim.x + threadIdx.x;
  size_t total = (size_t)K * N;
  if (idx >= total) return;
  int k = (int)(idx / N);
  int n = (int)(idx % N);
  Wt[(size_t)n * K + k] = (_Float16)W[idx];
}

// ---------------------------------------------------------------------------
// LayerNorm -> f16 activation (one block per row)
// ---------------------------------------------------------------------------
__global__ __launch_bounds__(256)
void layernorm_f16(const float* __restrict__ x, const float* __restrict__ w,
                   const float* __restrict__ b, _Float16* __restrict__ xn) {
  __shared__ float red[256];
  const int row = blockIdx.x;
  const float* xr = x + (size_t)row * DIM;

  float s = 0.f;
  for (int i = threadIdx.x; i < DIM; i += 256) s += xr[i];
  red[threadIdx.x] = s; __syncthreads();
  for (int off = 128; off > 0; off >>= 1) {
    if (threadIdx.x < off) red[threadIdx.x] += red[threadIdx.x + off];
    __syncthreads();
  }
  const float mu = red[0] * (1.0f / DIM);
  __syncthreads();

  float v = 0.f;
  for (int i = threadIdx.x; i < DIM; i += 256) {
    float d = xr[i] - mu; v += d * d;
  }
  red[threadIdx.x] = v; __syncthreads();
  for (int off = 128; off > 0; off >>= 1) {
    if (threadIdx.x < off) red[threadIdx.x] += red[threadIdx.x + off];
    __syncthreads();
  }
  const float rstd = rsqrtf(red[0] * (1.0f / DIM) + 1e-5f);

  for (int i = threadIdx.x; i < DIM; i += 256) {
    float val = (xr[i] - mu) * rstd * w[i] + b[i];
    xn[(size_t)row * DIM + i] = (_Float16)val;
  }
}

// ---------------------------------------------------------------------------
// Causal depthwise conv (DCONV=4) + bias + SiLU.  u = uz[:, 0:DI]
// ---------------------------------------------------------------------------
__global__ __launch_bounds__(256)
void conv_silu(const float* __restrict__ uz, const float* __restrict__ cw,
               const float* __restrict__ cb, float* __restrict__ uc,
               _Float16* __restrict__ uch) {
  size_t idx = (size_t)blockIdx.x * blockDim.x + threadIdx.x;
  if (idx >= (size_t)BL * DI) return;
  const int d = (int)(idx % DI);
  const size_t bl = idx / DI;
  const int l = (int)(bl % SEQ);
  const int bb = (int)(bl / SEQ);

  float acc = cb[d];
#pragma unroll
  for (int k = 0; k < DCONV; ++k) {
    int ls = l + k - (DCONV - 1);
    if (ls >= 0) {
      acc += uz[((size_t)bb * SEQ + ls) * (2 * DI) + d] * cw[d * DCONV + k];
    }
  }
  const float s = acc / (1.f + __expf(-acc));   // SiLU
  uc[idx] = s;
  uch[idx] = (_Float16)s;
}

// ---------------------------------------------------------------------------
// Misc elementwise kernels
// ---------------------------------------------------------------------------
__global__ __launch_bounds__(256)
void neg_exp_log2e(const float* __restrict__ A_log, float* __restrict__ negA, int n) {
  int i = blockIdx.x * blockDim.x + threadIdx.x;
  // -exp(A_log) * log2(e): scan then uses exp2(dt * a) == exp(dt * -exp(A_log))
  if (i < n) negA[i] = -__expf(A_log[i]) * 1.4426950408889634f;
}

__global__ __launch_bounds__(256)
void split_dtraw(const float* __restrict__ proj, _Float16* __restrict__ dtrh) {
  size_t idx = (size_t)blockIdx.x * blockDim.x + threadIdx.x;
  if (idx >= (size_t)BL * DTR) return;
  const size_t m = idx / DTR;
  const int c = (int)(idx % DTR);
  dtrh[idx] = (_Float16)proj[m * (DTR + 2 * NST) + c];
}

__global__ __launch_bounds__(256)
void softplus_bias(float* __restrict__ dt, const float* __restrict__ b_dt) {
  size_t idx = (size_t)blockIdx.x * blockDim.x + threadIdx.x;
  if (idx >= (size_t)BL * DI) return;
  const int d = (int)(idx % DI);
  const float v = dt[idx] + b_dt[d];
  dt[idx] = (v > 20.f) ? v : log1pf(__expf(v));
}

__global__ __launch_bounds__(256)
void gate_combine(const float* __restrict__ ysc, const float* __restrict__ uc,
                  const float* __restrict__ uz, const float* __restrict__ Dv,
                  _Float16* __restrict__ yh) {
  size_t idx = (size_t)blockIdx.x * blockDim.x + threadIdx.x;
  if (idx >= (size_t)BL * DI) return;
  const int d = (int)(idx % DI);
  const size_t row = idx / DI;
  const float z = uz[row * (2 * DI) + DI + d];
  const float sz = z / (1.f + __expf(-z));
  yh[idx] = (_Float16)((ysc[idx] + uc[idx] * Dv[d]) * sz);
}

// ---------------------------------------------------------------------------
// SSM selective scan: one thread per (batch, channel); h[64], A[64] in VGPRs.
// B_t / C_t (shared across channels) staged through LDS each step.
// grid = (DI/256, BATCH).  negA is pre-scaled by log2(e) -> use exp2.
// ---------------------------------------------------------------------------
__global__ __launch_bounds__(256)
void ssm_scan(const float* __restrict__ dt,    // [BL][DI]
              const float* __restrict__ proj,  // [BL][DTR+2*NST]
              const float* __restrict__ uc,    // [BL][DI]
              const float* __restrict__ negA,  // [DI][NST]  (pre-scaled)
              float* __restrict__ y) {         // [BL][DI]
  __shared__ float sB[NST];
  __shared__ float sC[NST];
  const int bb = blockIdx.y;
  const int d  = blockIdx.x * 256 + threadIdx.x;

  float h[NST];
  float a[NST];
#pragma unroll
  for (int n = 0; n < NST; ++n) {
    h[n] = 0.f;
    a[n] = negA[(size_t)d * NST + n];
  }

  for (int l = 0; l < SEQ; ++l) {
    const size_t row = (size_t)bb * SEQ + l;
    if (threadIdx.x < NST) {
      sB[threadIdx.x] = proj[row * (DTR + 2 * NST) + DTR + threadIdx.x];
    } else if (threadIdx.x < 2 * NST) {
      sC[threadIdx.x - NST] =
          proj[row * (DTR + 2 * NST) + DTR + NST + (threadIdx.x - NST)];
    }
    __syncthreads();

    const float dtv = dt[row * DI + d];
    const float xin = dtv * uc[row * DI + d];
    float yv = 0.f;
#pragma unroll
    for (int n = 0; n < NST; ++n) {
      const float dA = EXP2F(dtv * a[n]);   // v_exp_f32
      h[n] = h[n] * dA + xin * sB[n];
      yv += h[n] * sC[n];
    }
    y[row * DI + d] = yv;
    __syncthreads();
  }
}

// ---------------------------------------------------------------------------
// Host-side launch
// ---------------------------------------------------------------------------
static inline size_t alnup(size_t x) { return (x + 255) & ~size_t(255); }

extern "C" void kernel_launch(void* const* d_in, const int* in_sizes, int n_in,
                              void* d_out, int out_size, void* d_ws, size_t ws_size,
                              hipStream_t stream) {
  const float* x      = (const float*)d_in[0];
  const float* ln_w   = (const float*)d_in[1];
  const float* ln_b   = (const float*)d_in[2];
  const float* W_in   = (const float*)d_in[3];
  const float* conv_w = (const float*)d_in[4];
  const float* conv_b = (const float*)d_in[5];
  const float* W_xp   = (const float*)d_in[6];
  const float* W_dt   = (const float*)d_in[7];
  const float* b_dt   = (const float*)d_in[8];
  const float* A_log  = (const float*)d_in[9];
  const float* Dvec   = (const float*)d_in[10];
  const float* W_out  = (const float*)d_in[11];
  float* out = (float*)d_out;

  char* ws = (char*)d_ws;
  size_t off = 0;
  auto take = [&](size_t bytes) { char* p = ws + off; off += alnup(bytes); return p; };

  _Float16* xn_h   = (_Float16*)take((size_t)BL * DIM * 2);
  _Float16* WinT   = (_Float16*)take((size_t)(2 * DI) * DIM * 2);       // [4096][1024]
  float*    uz     = (float*)   take((size_t)BL * 2 * DI * 4);          // [8192][4096]
  float*    uc     = (float*)   take((size_t)BL * DI * 4);
  _Float16* uc_h   = (_Float16*)take((size_t)BL * DI * 2);
  _Float16* WxpT   = (_Float16*)take((size_t)(DTR + 2 * NST) * DI * 2); // [192][2048]
  float*    proj   = (float*)   take((size_t)BL * (DTR + 2 * NST) * 4);
  _Float16* dtr_h  = (_Float16*)take((size_t)BL * DTR * 2);
  _Float16* WdtT   = (_Float16*)take((size_t)DI * DTR * 2);             // [2048][64]
  float*    dtbuf  = (float*)   take((size_t)BL * DI * 4);
  float*    negA   = (float*)   take((size_t)DI * NST * 4);
  float*    ysc    = (float*)   take((size_t)BL * DI * 4);
  _Float16* y_h    = (_Float16*)take((size_t)BL * DI * 2);
  _Float16* WoutT  = (_Float16*)take((size_t)DIM * DI * 2);             // [1024][2048]

  // --- weight prep (f32 -> transposed f16) ---
  {
    size_t n;
    n = (size_t)DIM * (2 * DI);
    transpose_cvt_f16<<<dim3((unsigned)((n + 255) / 256)), 256, 0, stream>>>(W_in, WinT, DIM, 2 * DI);
    n = (size_t)DI * (DTR + 2 * NST);
    transpose_cvt_f16<<<dim3((unsigned)((n + 255) / 256)), 256, 0, stream>>>(W_xp, WxpT, DI, DTR + 2 * NST);
    n = (size_t)DTR * DI;
    transpose_cvt_f16<<<dim3((unsigned)((n + 255) / 256)), 256, 0, stream>>>(W_dt, WdtT, DTR, DI);
    n = (size_t)DI * DIM;
    transpose_cvt_f16<<<dim3((unsigned)((n + 255) / 256)), 256, 0, stream>>>(W_out, WoutT, DI, DIM);
    neg_exp_log2e<<<dim3((DI * NST + 255) / 256), 256, 0, stream>>>(A_log, negA, DI * NST);
  }

  // --- LayerNorm ---
  layernorm_f16<<<dim3(BL), 256, 0, stream>>>(x, ln_w, ln_b, xn_h);

  // --- GEMM1: xn @ W_in -> uz [8192 x 4096] ---
  gemm_wmma_f16<<<dim3((2 * DI) / 64, BL / 64), 128, 0, stream>>>(
      xn_h, WinT, nullptr, uz, BL, 2 * DI, DIM);

  // --- conv + SiLU ---
  conv_silu<<<dim3((unsigned)(((size_t)BL * DI + 255) / 256)), 256, 0, stream>>>(
      uz, conv_w, conv_b, uc, uc_h);

  // --- GEMM2: u_c @ W_xproj -> proj [8192 x 192] ---
  gemm_wmma_f16<<<dim3((DTR + 2 * NST) / 64, BL / 64), 128, 0, stream>>>(
      uc_h, WxpT, nullptr, proj, BL, DTR + 2 * NST, DI);

  // --- dt path ---
  split_dtraw<<<dim3((unsigned)(((size_t)BL * DTR + 255) / 256)), 256, 0, stream>>>(proj, dtr_h);
  gemm_wmma_f16<<<dim3(DI / 64, BL / 64), 128, 0, stream>>>(
      dtr_h, WdtT, nullptr, dtbuf, BL, DI, DTR);
  softplus_bias<<<dim3((unsigned)(((size_t)BL * DI + 255) / 256)), 256, 0, stream>>>(dtbuf, b_dt);

  // --- selective scan ---
  ssm_scan<<<dim3(DI / 256, BATCH), 256, 0, stream>>>(dtbuf, proj, uc, negA, ysc);

  // --- gate ---
  gate_combine<<<dim3((unsigned)(((size_t)BL * DI + 255) / 256)), 256, 0, stream>>>(
      ysc, uc, uz, Dvec, y_h);

  // --- GEMM4: y @ W_out + x -> out [8192 x 1024] ---
  gemm_wmma_f16<<<dim3(DIM / 64, BL / 64), 128, 0, stream>>>(
      y_h, WoutT, x, out, BL, DIM, DI);
}
